// MemoryModule_69466801045568
// MI455X (gfx1250) — matrix operand order repaired and checked by reading, not compiled, not run on hardware
//
#include <hip/hip_runtime.h>
#include <hip/hip_bf16.h>
#include <math.h>

// ---------------- problem constants (from reference) ----------------
#define BB   64
#define TT   512
#define HIDD 1024
#define MMEM 512     // mem slots (== TT)
#define NHD  8       // heads
#define HD   64      // head dim
#define NGG  1024    // num gates = 2*M

typedef _Float16 v16h __attribute__((ext_vector_type(16)));
typedef _Float16 v8h  __attribute__((ext_vector_type(8)));
typedef float    v8f  __attribute__((ext_vector_type(8)));

static __device__ __forceinline__ v8f wmma16(v16h a, v16h b, v8f c) {
  // D = A(16x32 f16) * B(32x16 f16) + C(16x16 f32)
  return __builtin_amdgcn_wmma_f32_16x16x32_f16(false, a, false, b, (short)0, c,
                                                false, false);
}
static __device__ __forceinline__ v16h frag_cat(v8h lo, v8h hi) {
  return __builtin_shufflevector(lo, hi, 0, 1, 2, 3, 4, 5, 6, 7, 8, 9, 10, 11,
                                 12, 13, 14, 15);
}
static __device__ __forceinline__ float sigmoidf_(float x) {
  return 1.0f / (1.0f + __expf(-x));
}

// =====================================================================
// Generic WMMA GEMM: C[rows x N] = epi( A[rows x K] @ W[K x N] + bias )
//   f32 operands in HBM, f16 multiply staged through double-buffered LDS,
//   f32 accumulate.  128x128 block tile, BK=32, 8 waves (4x2), each wave
//   owns a 2x4 grid of 16x16 WMMA tiles.
//   MODE 0: +bias            MODE 1: relu(+bias)
//   MODE 2: relu(colscale[n]*(acc+bias[n]))   (RepeatLinear gate path)
//   MODE 3: +bias + residual[row,n]           (fused "inputs + proj")
//   AOP  0: none             AOP 1: tanh(A) applied while staging
// rows, N, K all multiples of 128/128/32 for these shapes.
// =====================================================================
template <int MODE, int AOP>
__global__ __launch_bounds__(256) void gemm_wmma_kernel(
    const float* __restrict__ A, const float* __restrict__ W,
    const float* __restrict__ bias, const float* __restrict__ extra,
    float* __restrict__ C, int N, int K) {
  __shared__ __align__(16) _Float16 As[2][128 * 32];  // [row][k]
  __shared__ __align__(16) _Float16 Ws[2][128 * 32];  // [n][k] (transposed)
  const int tid = threadIdx.x;
  const int bm0 = blockIdx.y * 128;
  const int bn0 = blockIdx.x * 128;
  const int lane = tid & 31, wid = tid >> 5;
  const int wm = wid & 3, wn = wid >> 2;        // 4x2 wave grid
  const int fm = lane & 15, fg = lane >> 4;     // fragment lane coords

  v8f acc[2][4] = {};

  auto stage = [&](int k0, int p) {
    // ---- A tile (f32 -> f16), row-major [128][32] ----
#pragma unroll
    for (int i = tid * 4; i < 128 * 32; i += 1024) {
      const int r = i >> 5, c = i & 31;
      float4 f = *(const float4*)(A + (size_t)(bm0 + r) * K + (k0 + c));
      if (AOP == 1) {
        f.x = tanhf(f.x); f.y = tanhf(f.y); f.z = tanhf(f.z); f.w = tanhf(f.w);
      }
      As[p][i + 0] = (_Float16)f.x; As[p][i + 1] = (_Float16)f.y;
      As[p][i + 2] = (_Float16)f.z; As[p][i + 3] = (_Float16)f.w;
    }
    // ---- W tile transposed: Ws[n][k] ----
#pragma unroll
    for (int i = tid * 4; i < 32 * 128; i += 1024) {
      const int kk = i >> 7, c = i & 127;
      float4 f = *(const float4*)(W + (size_t)(k0 + kk) * N + (bn0 + c));
      Ws[p][(c + 0) * 32 + kk] = (_Float16)f.x;
      Ws[p][(c + 1) * 32 + kk] = (_Float16)f.y;
      Ws[p][(c + 2) * 32 + kk] = (_Float16)f.z;
      Ws[p][(c + 3) * 32 + kk] = (_Float16)f.w;
    }
  };

  stage(0, 0);
  __syncthreads();

  int p = 0;
  for (int k0 = 0; k0 < K; k0 += 32) {
    if (k0 + 32 < K) {
      stage(k0 + 32, p ^ 1);  // fill idle buffer while WMMAs run on buffer p
      if (k0 + 64 < K)        // hint tile after that -> global_prefetch_b8
        __builtin_prefetch(A + (size_t)(bm0 + (tid >> 1)) * K + k0 + 64, 0, 1);
    }

    v16h af[2], bf[4];
#pragma unroll
    for (int tm = 0; tm < 2; tm++) {
      const _Float16* pa = As[p] + (wm * 32 + tm * 16 + fm) * 32;
      af[tm] = frag_cat(*(const v8h*)(pa + 8 * fg), *(const v8h*)(pa + 16 + 8 * fg));
    }
#pragma unroll
    for (int tn = 0; tn < 4; tn++) {
      const _Float16* pb = Ws[p] + (wn * 64 + tn * 16 + fm) * 32;
      bf[tn] = frag_cat(*(const v8h*)(pb + 8 * fg), *(const v8h*)(pb + 16 + 8 * fg));
    }
#pragma unroll
    for (int tm = 0; tm < 2; tm++)
#pragma unroll
      for (int tn = 0; tn < 4; tn++) acc[tm][tn] = wmma16(af[tm], bf[tn], acc[tm][tn]);

    __syncthreads();  // staging for next step complete; buffer p free for reuse
    p ^= 1;
  }

  // ---- epilogue ----
#pragma unroll
  for (int tm = 0; tm < 2; tm++) {
#pragma unroll
    for (int tn = 0; tn < 4; tn++) {
      const int col = bn0 + wn * 64 + tn * 16 + fm;
      const float bcol = bias[col];
#pragma unroll
      for (int r = 0; r < 8; r++) {
        const int row = bm0 + wm * 32 + tm * 16 + r + 8 * fg;
        float v = acc[tm][tn][r] + bcol;
        if (MODE == 1) v = fmaxf(v, 0.0f);
        if (MODE == 2) v = fmaxf(extra[col] * v, 0.0f);
        if (MODE == 3) v += extra[(size_t)row * N + col];
        C[(size_t)row * N + col] = v;
      }
    }
  }
}

// =====================================================================
// Fused attention: per (batch, head, 16-query block) single wave.
//   scores = Q Kt / 8 (WMMA), branchless causal|ext mask (mask staged to
//   LDS as additive bias), softmax, exact top-8 threshold (8 argmax
//   passes, no renorm), attended = att @ V (WMMA).
// =====================================================================
__global__ __launch_bounds__(32) void attn_kernel(
    const float* __restrict__ Q, const float* __restrict__ Kb,
    const float* __restrict__ V, const unsigned char* __restrict__ amask,
    float* __restrict__ Out) {
  __shared__ float sc[16 * MMEM];                   // 32 KB scores
  __shared__ __align__(16) _Float16 ah[16 * MMEM];  // 16 KB att (f16)
  __shared__ __align__(16) _Float16 vt[HD * 32];    // 4 KB  V tile [d][key]
  __shared__ float mb[MMEM];                        // 2 KB  mask bias

  const int bx = blockIdx.x;
  const int qb = bx & 31;           // 16-row query block
  const int h = (bx >> 5) & 7;      // head
  const int b = bx >> 8;            // batch
  const int lane = threadIdx.x;
  const int fm = lane & 15, fg = lane >> 4;

  // ---- stage external mask once as additive bias (0 or -inf) ----
#pragma unroll
  for (int i = 0; i < MMEM / 32; i++) {
    const int key = lane + 32 * i;
    mb[key] = amask[b * MMEM + key] ? -INFINITY : 0.0f;
  }

  // ---- Q fragments (16 rows x 64 d) ----
  const float* qrow = Q + ((size_t)(b * TT + qb * 16 + fm)) * MMEM + h * HD;
  v16h qa[2];
#pragma unroll
  for (int c = 0; c < 2; c++) {
    v16h a;
#pragma unroll
    for (int e = 0; e < 8; e++) a[e] = (_Float16)qrow[c * 32 + 8 * fg + e];
#pragma unroll
    for (int e = 0; e < 8; e++) a[8 + e] = (_Float16)qrow[c * 32 + 16 + 8 * fg + e];
    qa[c] = a;
  }
  __syncthreads();

  // ---- scores ----
  const float* krow = Kb + ((size_t)(b * TT)) * MMEM + h * HD;
  for (int kt = 0; kt < 32; kt++) {
    v8f acc = {};
    const float* kr = krow + (size_t)(kt * 16 + fm) * MMEM;
#pragma unroll
    for (int c = 0; c < 2; c++) {
      v16h bb;
#pragma unroll
      for (int e = 0; e < 16; e++) bb[e] = (_Float16)kr[c * 32 + 8 * fg + e + (e >= 8 ? 8 : 0)];
      acc = wmma16(qa[c], bb, acc);
    }
    const int key = kt * 16 + fm;
    const float mbias = mb[key];
#pragma unroll
    for (int r = 0; r < 8; r++) {
      const int row = r + 8 * fg;
      const int qi = qb * 16 + row;
      const float s = fmaf(acc[r], 0.125f, mbias);     // ext mask folded in
      sc[row * MMEM + key] = (key > qi) ? -INFINITY : s;  // causal: cndmask
    }
  }
  __syncthreads();

  // ---- per-row softmax + exact top-8 threshold ----
  for (int r = 0; r < 16; r++) {
    float vals[16];
#pragma unroll
    for (int i = 0; i < 16; i++) vals[i] = sc[r * MMEM + lane + 32 * i];
    float mx = -INFINITY;
#pragma unroll
    for (int i = 0; i < 16; i++) mx = fmaxf(mx, vals[i]);
    for (int o = 16; o >= 1; o >>= 1) mx = fmaxf(mx, __shfl_xor(mx, o, 32));
    float sum = 0.0f;
#pragma unroll
    for (int i = 0; i < 16; i++) sum += __expf(vals[i] - mx);
    for (int o = 16; o >= 1; o >>= 1) sum += __shfl_xor(sum, o, 32);
    // 8 argmax passes -> 8th largest score (ties handled like sorted order)
    float t[16];
#pragma unroll
    for (int i = 0; i < 16; i++) t[i] = vals[i];
    float s8 = -INFINITY;
    for (int it = 0; it < 8; it++) {
      float lm = -INFINITY; int li = -1;
#pragma unroll
      for (int i = 0; i < 16; i++)
        if (t[i] > lm) { lm = t[i]; li = i; }
      float bm = lm; int bl = lane;
      for (int o = 16; o >= 1; o >>= 1) {
        float om = __shfl_xor(bm, o, 32);
        int ol = __shfl_xor(bl, o, 32);
        if (om > bm || (om == bm && ol < bl)) { bm = om; bl = ol; }
      }
      s8 = bm;
      if (bl == lane && li >= 0) t[li] = -INFINITY;
    }
    const float inv = 1.0f / sum;
#pragma unroll
    for (int i = 0; i < 16; i++) {
      const float s = vals[i];
      const float p = (s >= s8) ? __expf(s - mx) * inv : 0.0f;
      ah[r * MMEM + lane + 32 * i] = (_Float16)p;
    }
  }
  __syncthreads();

  // ---- attended = att @ V ----
  v8f accD[4] = {};
  for (int kt2 = 0; kt2 < 16; kt2++) {
    // stage V tile (32 keys x 64 d) transposed into vt[d][key]
    const float* vr = V + ((size_t)(b * TT + kt2 * 32 + lane)) * MMEM + h * HD;
#pragma unroll
    for (int d = 0; d < HD; d++) vt[d * 32 + lane] = (_Float16)vr[d];
    __syncthreads();

    const _Float16* ab = ah + fm * MMEM + kt2 * 32;
    v16h af = frag_cat(*(const v8h*)(ab + 8 * fg), *(const v8h*)(ab + 16 + 8 * fg));
#pragma unroll
    for (int nt = 0; nt < 4; nt++) {
      const _Float16* bb = vt + (nt * 16 + fm) * 32 + 8 * fg;
      v16h bf = frag_cat(*(const v8h*)bb, *(const v8h*)(bb + 16));
      accD[nt] = wmma16(af, bf, accD[nt]);
    }
    __syncthreads();
  }
#pragma unroll
  for (int nt = 0; nt < 4; nt++)
#pragma unroll
    for (int r = 0; r < 8; r++) {
      const int row = qb * 16 + r + 8 * fg;
      const int col = h * HD + nt * 16 + fm;
      Out[((size_t)(b * TT + row)) * MMEM + col] = accD[nt][r];
    }
}

// ---------------- LayerNorm over 512 cols with fused residual ----------------
__global__ __launch_bounds__(256) void ln_add_kernel(
    const float* __restrict__ X, const float* __restrict__ Yadd,
    const float* __restrict__ g, const float* __restrict__ bta,
    float* __restrict__ Out) {
  __shared__ float red[256];
  const int tid = threadIdx.x;
  const size_t row = blockIdx.x;
  const float* x = X + row * MMEM;
  const float* y = Yadd + row * MMEM;
  const float v0 = x[tid] + y[tid];
  const float v1 = x[tid + 256] + y[tid + 256];
  red[tid] = v0 + v1;
  __syncthreads();
  for (int o = 128; o > 0; o >>= 1) {
    if (tid < o) red[tid] += red[tid + o];
    __syncthreads();
  }
  const float mu = red[0] * (1.0f / MMEM);
  __syncthreads();
  const float d0 = v0 - mu, d1 = v1 - mu;
  red[tid] = d0 * d0 + d1 * d1;
  __syncthreads();
  for (int o = 128; o > 0; o >>= 1) {
    if (tid < o) red[tid] += red[tid + o];
    __syncthreads();
  }
  const float rs = rsqrtf(red[0] * (1.0f / MMEM) + 1e-5f);
  Out[row * MMEM + tid] = d0 * rs * g[tid] + bta[tid];
  Out[row * MMEM + tid + 256] = d1 * rs * g[tid + 256] + bta[tid + 256];
}

// ------------- mean over T of P[b,t,m] (P already relu/scaled) -------------
__global__ void mean_t_kernel(const float* __restrict__ P, float* __restrict__ xm) {
  const int cidx = blockIdx.x * blockDim.x + threadIdx.x;  // b*M + m
  const int b = cidx >> 9, m = cidx & 511;
  const float* p = P + (size_t)b * TT * MMEM + m;
  float s = 0.0f;
  for (int t = 0; t < TT; t++) s += p[(size_t)t * MMEM];
  xm[cidx] = s * (1.0f / TT);
}

// ------------- gate_inputs[b, j] = xm[b,:] @ rl_W + rl_b  (tiny) -------------
__global__ void gate_in_kernel(const float* __restrict__ xm,
                               const float* __restrict__ rlW,
                               const float* __restrict__ rlb,
                               float* __restrict__ gin) {
  const int idx = blockIdx.x * blockDim.x + threadIdx.x;  // b*NG + j
  const int b = idx >> 10, j = idx & 1023;
  const float* x = xm + b * MMEM;
  float s = 0.0f;
  for (int m = 0; m < MMEM; m++) s += x[m] * rlW[(size_t)m * NGG + j];
  gin[idx] = s + rlb[j];
}

// --------- final gating: next_memory = ig*tanh(nm) + fg*memory ---------
__global__ void combine_kernel(const float* __restrict__ gmem,
                               const float* __restrict__ gin,
                               const float* __restrict__ nm,
                               const float* __restrict__ mem,
                               const float* __restrict__ ibias,
                               const float* __restrict__ fbias,
                               float* __restrict__ out2) {
  const size_t idx = (size_t)blockIdx.x * blockDim.x + threadIdx.x;
  const int b = (int)(idx >> 18);
  const int rem = (int)(idx & 262143);
  const int i = rem >> 9, j = rem & 511;
  const size_t grow = ((size_t)(b * MMEM + i)) * NGG;
  const float gi = gmem[grow + j] + gin[b * NGG + j] + ibias[0];
  const float gf = gmem[grow + MMEM + j] + gin[b * NGG + MMEM + j] + fbias[0];
  out2[idx] = sigmoidf_(gi) * tanhf(nm[idx]) + sigmoidf_(gf) * mem[idx];
}

// =====================================================================
extern "C" void kernel_launch(void* const* d_in, const int* in_sizes, int n_in,
                              void* d_out, int out_size, void* d_ws,
                              size_t ws_size, hipStream_t stream) {
  const float* inputs = (const float*)d_in[0];
  const float* memory = (const float*)d_in[1];
  const unsigned char* amask = (const unsigned char*)d_in[2];
  const float* Wq = (const float*)d_in[3];   const float* bq = (const float*)d_in[4];
  const float* Wk = (const float*)d_in[5];   const float* bk = (const float*)d_in[6];
  const float* Wv = (const float*)d_in[7];   const float* bv = (const float*)d_in[8];
  const float* Wmlp = (const float*)d_in[9]; const float* bmlp = (const float*)d_in[10];
  const float* ln1g = (const float*)d_in[11]; const float* ln1b = (const float*)d_in[12];
  const float* ln2g = (const float*)d_in[13]; const float* ln2b = (const float*)d_in[14];
  const float* rl_w = (const float*)d_in[15];
  const float* rl_pW = (const float*)d_in[16]; const float* rl_pb = (const float*)d_in[17];
  const float* rl_W = (const float*)d_in[18];  const float* rl_b = (const float*)d_in[19];
  const float* mg_W = (const float*)d_in[20];  const float* mg_b = (const float*)d_in[21];
  const float* out_W = (const float*)d_in[22]; const float* out_b = (const float*)d_in[23];
  const float* ibias = (const float*)d_in[24];
  const float* fbias = (const float*)d_in[25];

  float* out1 = (float*)d_out;                           // [B,T,HID]
  float* out2 = out1 + (size_t)BB * TT * HIDD;           // [B,M,M] next_memory

  char* w = (char*)d_ws;  // overlapped workspace (~321 MB)
  float* P    = (float*)(w + 0);           // 64MB: gate-proj P, later attended
  float* ATT  = P;                         //   (P consumed before attention)
  float* MEM2 = (float*)(w + 67108864);    // 64MB
  float* GMEM = (float*)(w + 0);           // 128MB (P/ATT/MEM2 dead by then)
  float* Qb   = (float*)(w + 134217728);   // 64MB, later H1
  float* Kb   = (float*)(w + 201326592);   // 64MB, later H2
  float* Vb   = (float*)(w + 268435456);   // 64MB, later next_mem (LN2 out)
  float* H1 = Qb;  float* H2 = Kb;  float* NM = Vb;
  float* XM  = (float*)(w + 335544320);    // [B,M]
  float* GIN = (float*)(w + 335675392);    // [B,NG]

  const int ROWS = BB * TT;  // 32768 (== B*M)
  dim3 blk(256);

  // 1) P = relu(rl_w * (inputs @ rl_pW + rl_pb))   [ROWS x 512], K=1024
  gemm_wmma_kernel<2, 0><<<dim3(MMEM / 128, ROWS / 128), blk, 0, stream>>>(
      inputs, rl_pW, rl_pb, rl_w, P, MMEM, HIDD);
  // 2) xm = mean_T(P);  3) gate_inputs
  mean_t_kernel<<<dim3(BB * MMEM / 256), blk, 0, stream>>>(P, XM);
  gate_in_kernel<<<dim3(BB * NGG / 256), blk, 0, stream>>>(XM, rl_W, rl_b, GIN);

  // 4-6) q/k/v projections
  gemm_wmma_kernel<0, 0><<<dim3(MMEM / 128, ROWS / 128), blk, 0, stream>>>(
      inputs, Wq, bq, nullptr, Qb, MMEM, HIDD);
  gemm_wmma_kernel<0, 0><<<dim3(MMEM / 128, ROWS / 128), blk, 0, stream>>>(
      memory, Wk, bk, nullptr, Kb, MMEM, MMEM);
  gemm_wmma_kernel<0, 0><<<dim3(MMEM / 128, ROWS / 128), blk, 0, stream>>>(
      memory, Wv, bv, nullptr, Vb, MMEM, MMEM);

  // 7) fused attention -> ATT (reuses P region)
  attn_kernel<<<dim3(BB * NHD * (TT / 16)), dim3(32), 0, stream>>>(Qb, Kb, Vb,
                                                                   amask, ATT);

  // 8) mem2 = LN(memory + attended)
  ln_add_kernel<<<dim3(ROWS), blk, 0, stream>>>(memory, ATT, ln1g, ln1b, MEM2);

  // 9-10) shared 2-layer relu MLP (same weights twice)
  gemm_wmma_kernel<1, 0><<<dim3(MMEM / 128, ROWS / 128), blk, 0, stream>>>(
      MEM2, Wmlp, bmlp, nullptr, H1, MMEM, MMEM);
  gemm_wmma_kernel<1, 0><<<dim3(MMEM / 128, ROWS / 128), blk, 0, stream>>>(
      H1, Wmlp, bmlp, nullptr, H2, MMEM, MMEM);

  // 11) next_mem = LN(mem2 + h)
  ln_add_kernel<<<dim3(ROWS), blk, 0, stream>>>(MEM2, H2, ln2g, ln2b, NM);

  // 12) gate_memory = tanh(memory) @ mg_W + mg_b   [ROWS x 1024]
  gemm_wmma_kernel<0, 1><<<dim3(NGG / 128, ROWS / 128), blk, 0, stream>>>(
      memory, mg_W, mg_b, nullptr, GMEM, NGG, MMEM);

  // 13) next_memory (output 2)
  combine_kernel<<<dim3((BB * MMEM * MMEM) / 256), blk, 0, stream>>>(
      GMEM, GIN, NM, memory, ibias, fbias, out2);

  // 14) output 1 = inputs + next_memory @ out_W + out_b  (fused residual)
  gemm_wmma_kernel<3, 0><<<dim3(HIDD / 128, ROWS / 128), blk, 0, stream>>>(
      out2, out_W, out_b, inputs, out1, HIDD, MMEM);
}
